// CondorMoELayer_11544872091891
// MI455X (gfx1250) — matrix operand order repaired and checked by reading, not compiled, not run on hardware
//
#include <hip/hip_runtime.h>
#include <hip/hip_bf16.h>
#include <math.h>

#define T_TOKENS 4096
#define H_DIM    1024
#define I_DIM    4096
#define NEXP     8
#define MTILE    128
#define NTILE    128
#define KBLK     64
#define LDS_STRIDE (KBLK + 8)          // 72 bf16 = 144B row stride (16B aligned, conflict-free)
#define MAX_MT   (T_TOKENS / MTILE)    // 32 worst-case M tiles per expert

typedef __bf16 bf16;
typedef unsigned int       u32;
typedef unsigned long long u64;
typedef __attribute__((ext_vector_type(16))) __bf16 v16bf;
typedef __attribute__((ext_vector_type(8)))  __bf16 v8bf;
typedef __attribute__((ext_vector_type(8)))  float  v8f;
typedef __attribute__((ext_vector_type(4)))  u32    v4u32;
typedef __attribute__((ext_vector_type(8)))  int    v8i32;
typedef __attribute__((ext_vector_type(4)))  int    v4i32;

// ---------------- fragment loaders (per CDNA5 ISA 16-bit WMMA layouts) ----------------
// A (16x32 bf16): m = lane&15; lanes 0-15 hold K 0-7 & 16-23, lanes 16-31 hold K 8-15 & 24-31.
__device__ inline v16bf load_frag_a(const bf16* base, int lane) {
    int m   = lane & 15;
    int klo = (lane >> 4) << 3;                     // 0 or 8
    const bf16* p = base + m * LDS_STRIDE + klo;
    v8bf lo = *(const v8bf*)(p);                    // K = klo .. klo+7
    v8bf hi = *(const v8bf*)(p + 16);               // K = klo+16 .. klo+23
    return __builtin_shufflevector(lo, hi, 0,1,2,3,4,5,6,7,8,9,10,11,12,13,14,15);
}
// B (32x16 bf16): n = lane&15; lanes 0-15 hold K 0-7 & 16-23, lanes 16-31 hold K 8-15 & 24-31.
__device__ inline v16bf load_frag_b(const bf16* base, int lane) {
    int n  = lane & 15;
    int kh = (lane >> 4) << 3;                      // 0 or 8
    const bf16* p = base + n * LDS_STRIDE + kh;
    v8bf lo = *(const v8bf*)(p);
    v8bf hi = *(const v8bf*)(p + 16);
    return __builtin_shufflevector(lo, hi, 0,1,2,3,4,5,6,7,8,9,10,11,12,13,14,15);
}

// ---------------- TDM: async 2D tile load (128 rows x 64 bf16) global -> LDS ----------------
// Reproduces the padded LDS layout: 128B data rows + 16B pad (pad_interval=32 DW, pad_amount=4 DW).
__device__ inline void tdm_load_tile_bf16(const bf16* gptr, u32 lds_off) {
#if __has_builtin(__builtin_amdgcn_tensor_load_to_lds)
    u64 ga = (u64)(uintptr_t)gptr;
    v4u32 g0;
    g0[0] = 1u;                                        // count=1, user mode
    g0[1] = lds_off;                                   // LDS byte address
    g0[2] = (u32)(ga & 0xffffffffu);                   // global_addr[31:0]
    g0[3] = (u32)((ga >> 32) & 0x01ffffffu) | (2u << 30); // global_addr[56:32], type=2 (image)
    v8i32 g1;
    g1[0] = (1 << 16) | (1 << 20) | (4 << 22) | (3 << 25); // data_size=2B, pad_en, int=32DW, amt=4DW
    g1[1] = (64 << 16);                                // tensor_dim0[15:0] = 64
    g1[2] = (128 << 16);                               // tensor_dim0 hi=0 | tensor_dim1[15:0]=128
    g1[3] = (64 << 16);                                // tensor_dim1 hi=0 | tile_dim0=64
    g1[4] = 128;                                       // tile_dim1=128, tile_dim2=0
    g1[5] = I_DIM;                                     // tensor_dim0_stride = 4096 elems
    g1[6] = 0;
    g1[7] = 0;
    v4i32 z4 = {0, 0, 0, 0};
    v8i32 z8 = {0, 0, 0, 0, 0, 0, 0, 0};
    __builtin_amdgcn_tensor_load_to_lds(g0, g1, z4, z4, z8, 0);   // 6-arg (clang-23) form
#endif
}

// ---------------- router: one wave32 per token ----------------
__global__ __launch_bounds__(256)
void router_kernel(const float* __restrict__ x, const float* __restrict__ rw,
                   int* __restrict__ counts, int* __restrict__ tok_idx,
                   float* __restrict__ tok_w) {
    int wave = threadIdx.x >> 5;
    int lane = threadIdx.x & 31;
    int t = blockIdx.x * 8 + wave;
    if (t >= T_TOKENS) return;
    const float* xt = x + (size_t)t * H_DIM;
    float acc[NEXP];
#pragma unroll
    for (int e = 0; e < NEXP; ++e) acc[e] = 0.f;
    for (int h = lane; h < H_DIM; h += 32) {
        float xv = xt[h];
#pragma unroll
        for (int e = 0; e < NEXP; ++e)
            acc[e] = fmaf(xv, rw[e * H_DIM + h], acc[e]);
    }
#pragma unroll
    for (int e = 0; e < NEXP; ++e) {
#pragma unroll
        for (int off = 16; off > 0; off >>= 1)
            acc[e] += __shfl_xor(acc[e], off, 32);
    }
    if (lane == 0) {
        float mx = acc[0];
#pragma unroll
        for (int e = 1; e < NEXP; ++e) mx = fmaxf(mx, acc[e]);
        float p[NEXP]; float s = 0.f;
#pragma unroll
        for (int e = 0; e < NEXP; ++e) { p[e] = __expf(acc[e] - mx); s += p[e]; }
        float inv_s = 1.f / s;
#pragma unroll
        for (int e = 0; e < NEXP; ++e) p[e] *= inv_s;
        int i1 = 0;
#pragma unroll
        for (int e = 1; e < NEXP; ++e) if (p[e] > p[i1]) i1 = e;
        int i2 = (i1 == 0) ? 1 : 0;
#pragma unroll
        for (int e = 0; e < NEXP; ++e) if (e != i1 && p[e] > p[i2]) i2 = e;
        float w1 = p[i1], w2 = p[i2];
        float inv = 1.f / (w1 + w2);
        w1 *= inv; w2 *= inv;
        int pos1 = atomicAdd(&counts[i1], 1);
        tok_idx[i1 * T_TOKENS + pos1] = t;
        tok_w  [i1 * T_TOKENS + pos1] = w1;
        int pos2 = atomicAdd(&counts[i2], 1);
        tok_idx[i2 * T_TOKENS + pos2] = t;
        tok_w  [i2 * T_TOKENS + pos2] = w2;
    }
}

// ---------------- tiny scan: padded row offsets for compact intermediate ----------------
__global__ void scan_kernel(const int* __restrict__ counts, int* __restrict__ row_off) {
    if (threadIdx.x == 0) {
        int off = 0;
        for (int e = 0; e < NEXP; ++e) {
            row_off[e] = off;
            off += ((counts[e] + MTILE - 1) / MTILE) * MTILE;
        }
    }
}

// ---------------- pass A: inter = gelu(gather(X) @ W_in[e] + b_in[e]) ----------------
__global__ __launch_bounds__(256)
void expert_in_kernel(const float* __restrict__ x, const float* __restrict__ w_in,
                      const float* __restrict__ b_in, const int* __restrict__ counts,
                      const int* __restrict__ row_off, const int* __restrict__ tok_idx,
                      bf16* __restrict__ inter) {
    __shared__ bf16 lA[MTILE * LDS_STRIDE];
    __shared__ bf16 lB[NTILE * LDS_STRIDE];
    int e  = blockIdx.x / MAX_MT;
    int mt = blockIdx.x % MAX_MT;
    int cnt = counts[e];
    if (mt * MTILE >= cnt) return;
    int n0 = blockIdx.y * NTILE;
    const float* W = w_in + (size_t)e * H_DIM * I_DIM;
    const int* idx = tok_idx + e * T_TOKENS + mt * MTILE;

    int tid = threadIdx.x;
    int wave = tid >> 5, lane = tid & 31;
    int wm = (wave >> 2) * 64;     // 0 / 64
    int wn = (wave & 3) * 32;      // 0 / 32 / 64 / 96

    // hoisted gather row pointers for this thread's 8 A-chunks
    const float* arow[8];
#pragma unroll
    for (int i = 0; i < 8; ++i) {
        int id = tid + 256 * i;
        int r  = id >> 4;
        arow[i] = (mt * MTILE + r < cnt) ? (x + (size_t)idx[r] * H_DIM) : nullptr;
    }

    v8f acc[4][2];
#pragma unroll
    for (int mi = 0; mi < 4; ++mi)
#pragma unroll
        for (int ni = 0; ni < 2; ++ni) acc[mi][ni] = v8f{};

    float4 aSt[8], bSt[8];
    // prologue: stage tile k0 = 0
#pragma unroll
    for (int i = 0; i < 8; ++i) {
        int id = tid + 256 * i;
        int c  = (id & 15) << 2;
        aSt[i] = arow[i] ? *(const float4*)(arow[i] + c) : make_float4(0.f, 0.f, 0.f, 0.f);
    }
#pragma unroll
    for (int i = 0; i < 8; ++i) {
        int id = tid + 256 * i;
        int k  = id >> 5;
        int c  = (id & 31) << 2;
        bSt[i] = *(const float4*)(W + (size_t)k * I_DIM + n0 + c);
    }

    for (int k0 = 0; k0 < H_DIM; k0 += KBLK) {
        __syncthreads();
        // commit staged tiles to LDS (fp32 -> bf16)
#pragma unroll
        for (int i = 0; i < 8; ++i) {
            int id = tid + 256 * i;
            int r  = id >> 4;
            int c  = (id & 15) << 2;
            bf16* dst = lA + r * LDS_STRIDE + c;
            dst[0] = (bf16)aSt[i].x; dst[1] = (bf16)aSt[i].y;
            dst[2] = (bf16)aSt[i].z; dst[3] = (bf16)aSt[i].w;
        }
#pragma unroll
        for (int i = 0; i < 8; ++i) {
            int id = tid + 256 * i;
            int k  = id >> 5;
            int c  = (id & 31) << 2;
            lB[(c + 0) * LDS_STRIDE + k] = (bf16)bSt[i].x;
            lB[(c + 1) * LDS_STRIDE + k] = (bf16)bSt[i].y;
            lB[(c + 2) * LDS_STRIDE + k] = (bf16)bSt[i].z;
            lB[(c + 3) * LDS_STRIDE + k] = (bf16)bSt[i].w;
        }
        int kn = k0 + KBLK;
        if (kn < H_DIM) {
            // stage next tile while this one is consumed
#pragma unroll
            for (int i = 0; i < 8; ++i) {
                int id = tid + 256 * i;
                int c  = (id & 15) << 2;
                aSt[i] = arow[i] ? *(const float4*)(arow[i] + kn + c)
                                 : make_float4(0.f, 0.f, 0.f, 0.f);
            }
#pragma unroll
            for (int i = 0; i < 8; ++i) {
                int id = tid + 256 * i;
                int k  = id >> 5;
                int c  = (id & 31) << 2;
                bSt[i] = *(const float4*)(W + (size_t)(kn + k) * I_DIM + n0 + c);
            }
            if (kn + KBLK < H_DIM)  // L2 prefetch of the K+2 weight block
                __builtin_prefetch(W + (size_t)(kn + KBLK + (tid >> 2)) * I_DIM + n0 + ((tid & 3) << 5), 0, 0);
        }
        __syncthreads();
#pragma unroll
        for (int kk = 0; kk < KBLK; kk += 32) {
            v16bf afr[4], bfr[2];
#pragma unroll
            for (int mi = 0; mi < 4; ++mi)
                afr[mi] = load_frag_a(lA + (wm + mi * 16) * LDS_STRIDE + kk, lane);
#pragma unroll
            for (int ni = 0; ni < 2; ++ni)
                bfr[ni] = load_frag_b(lB + (wn + ni * 16) * LDS_STRIDE + kk, lane);
#pragma unroll
            for (int mi = 0; mi < 4; ++mi)
#pragma unroll
                for (int ni = 0; ni < 2; ++ni)
                    acc[mi][ni] = __builtin_amdgcn_wmma_f32_16x16x32_bf16(
                        false, afr[mi], false, bfr[ni], (short)0, acc[mi][ni], false, false);
        }
    }
    // epilogue: bias + exact GELU, store bf16 intermediate (compact rows, zero padding rows)
    int rbase = row_off[e] + mt * MTILE;
    int lane_n = lane & 15;
    int mhalf  = (lane >> 4) << 3;
#pragma unroll
    for (int mi = 0; mi < 4; ++mi) {
#pragma unroll
        for (int ni = 0; ni < 2; ++ni) {
            int ng = n0 + wn + ni * 16 + lane_n;
            float bias = b_in[e * I_DIM + ng];
#pragma unroll
            for (int r = 0; r < 8; ++r) {
                int ml   = wm + mi * 16 + r + mhalf;
                int slot = mt * MTILE + ml;
                float v = acc[mi][ni][r] + bias;
                v = 0.5f * v * (1.f + erff(v * 0.70710678118f));
                if (slot >= cnt) v = 0.f;
                inter[(size_t)(rbase + ml) * I_DIM + ng] = (bf16)v;
            }
        }
    }
}

// ---------------- pass B: out[token] += w * (inter @ W_out[e] + b_out[e]) ----------------
__global__ __launch_bounds__(256)
void expert_out_kernel(const bf16* __restrict__ inter, const float* __restrict__ w_out,
                       const float* __restrict__ b_out, const int* __restrict__ counts,
                       const int* __restrict__ row_off, const int* __restrict__ tok_idx,
                       const float* __restrict__ tok_w, float* __restrict__ out) {
    __shared__ bf16 lA2[2][MTILE * LDS_STRIDE];   // double-buffered: filled by TDM
    __shared__ bf16 lB[NTILE * LDS_STRIDE];
    int e  = blockIdx.x / MAX_MT;
    int mt = blockIdx.x % MAX_MT;
    int cnt = counts[e];
    if (mt * MTILE >= cnt) return;
    int n0 = blockIdx.y * NTILE;
    const float* W = w_out + (size_t)e * I_DIM * H_DIM;
    int rbase = row_off[e] + mt * MTILE;
    const bf16* Abase = inter + (size_t)rbase * I_DIM;

    int tid = threadIdx.x;
    int wave = tid >> 5, lane = tid & 31;
    int wm = (wave >> 2) * 64;
    int wn = (wave & 3) * 32;

    v8f acc[4][2];
#pragma unroll
    for (int mi = 0; mi < 4; ++mi)
#pragma unroll
        for (int ni = 0; ni < 2; ++ni) acc[mi][ni] = v8f{};

    float4 bSt[8];
#if __has_builtin(__builtin_amdgcn_tensor_load_to_lds)
    if (tid == 0)
        tdm_load_tile_bf16(Abase, (u32)(uintptr_t)&lA2[0][0]);   // async A tile k0=0
#endif
#pragma unroll
    for (int i = 0; i < 8; ++i) {
        int id = tid + 256 * i;
        int k  = id >> 5;
        int c  = (id & 31) << 2;
        bSt[i] = *(const float4*)(W + (size_t)k * H_DIM + n0 + c);
    }

    int buf = 0;
    for (int k0 = 0; k0 < I_DIM; k0 += KBLK) {
#if __has_builtin(__builtin_amdgcn_tensor_load_to_lds)
        __builtin_amdgcn_s_wait_tensorcnt(0);   // A tile for this iteration landed in LDS
#endif
        __syncthreads();
#if !__has_builtin(__builtin_amdgcn_tensor_load_to_lds)
        // fallback: manual A tile copy (bf16, 16B chunks)
#pragma unroll
        for (int i = 0; i < 4; ++i) {
            int id = tid + 256 * i;
            int r  = id >> 3;
            int c  = (id & 7) << 3;
            *(uint4*)(&lA2[buf][0] + r * LDS_STRIDE + c) =
                *(const uint4*)(Abase + (size_t)r * I_DIM + k0 + c);
        }
#endif
        // commit staged B tile (fp32 -> bf16, transposed)
#pragma unroll
        for (int i = 0; i < 8; ++i) {
            int id = tid + 256 * i;
            int k  = id >> 5;
            int c  = (id & 31) << 2;
            lB[(c + 0) * LDS_STRIDE + k] = (bf16)bSt[i].x;
            lB[(c + 1) * LDS_STRIDE + k] = (bf16)bSt[i].y;
            lB[(c + 2) * LDS_STRIDE + k] = (bf16)bSt[i].z;
            lB[(c + 3) * LDS_STRIDE + k] = (bf16)bSt[i].w;
        }
        int kn = k0 + KBLK;
        if (kn < I_DIM) {
#if __has_builtin(__builtin_amdgcn_tensor_load_to_lds)
            if (tid == 0)   // async-load next A tile into the other buffer during compute
                tdm_load_tile_bf16(Abase + kn, (u32)(uintptr_t)&lA2[buf ^ 1][0]);
#endif
#pragma unroll
            for (int i = 0; i < 8; ++i) {
                int id = tid + 256 * i;
                int k  = id >> 5;
                int c  = (id & 31) << 2;
                bSt[i] = *(const float4*)(W + (size_t)(kn + k) * H_DIM + n0 + c);
            }
            if (kn + KBLK < I_DIM)
                __builtin_prefetch(W + (size_t)(kn + KBLK + (tid >> 2)) * H_DIM + n0 + ((tid & 3) << 5), 0, 0);
        }
        __syncthreads();
#pragma unroll
        for (int kk = 0; kk < KBLK; kk += 32) {
            v16bf afr[4], bfr[2];
#pragma unroll
            for (int mi = 0; mi < 4; ++mi)
                afr[mi] = load_frag_a(&lA2[buf][0] + (wm + mi * 16) * LDS_STRIDE + kk, lane);
#pragma unroll
            for (int ni = 0; ni < 2; ++ni)
                bfr[ni] = load_frag_b(lB + (wn + ni * 16) * LDS_STRIDE + kk, lane);
#pragma unroll
            for (int mi = 0; mi < 4; ++mi)
#pragma unroll
                for (int ni = 0; ni < 2; ++ni)
                    acc[mi][ni] = __builtin_amdgcn_wmma_f32_16x16x32_bf16(
                        false, afr[mi], false, bfr[ni], (short)0, acc[mi][ni], false, false);
        }
        buf ^= 1;
    }
    // epilogue: bias, scale by routing weight, scatter-add into output
    const int* idx = tok_idx + e * T_TOKENS + mt * MTILE;
    const float* wgt = tok_w + e * T_TOKENS + mt * MTILE;
    int lane_n = lane & 15;
    int mhalf  = (lane >> 4) << 3;
#pragma unroll
    for (int mi = 0; mi < 4; ++mi) {
#pragma unroll
        for (int ni = 0; ni < 2; ++ni) {
            int ng = n0 + wn + ni * 16 + lane_n;
            float bias = b_out[e * H_DIM + ng];
#pragma unroll
            for (int r = 0; r < 8; ++r) {
                int ml   = wm + mi * 16 + r + mhalf;
                int slot = mt * MTILE + ml;
                if (slot < cnt) {
                    int tok  = idx[ml];
                    float sc = wgt[ml];
                    atomicAdd(&out[(size_t)tok * H_DIM + ng], (acc[mi][ni][r] + bias) * sc);
                }
            }
        }
    }
}

extern "C" void kernel_launch(void* const* d_in, const int* in_sizes, int n_in,
                              void* d_out, int out_size, void* d_ws, size_t ws_size,
                              hipStream_t stream) {
    const float* x        = (const float*)d_in[0];
    const float* router_w = (const float*)d_in[1];
    const float* w_in     = (const float*)d_in[2];
    const float* b_in     = (const float*)d_in[3];
    const float* w_out    = (const float*)d_in[4];
    const float* b_out    = (const float*)d_in[5];
    float* out = (float*)d_out;

    char* ws = (char*)d_ws;
    int*   counts  = (int*)ws;                                      // 8 ints
    int*   row_off = (int*)(ws + 64);                               // 8 ints
    int*   tok_idx = (int*)(ws + 128);                              // 8*4096 ints
    float* tok_w   = (float*)(ws + 128 + NEXP * T_TOKENS * 4);      // 8*4096 floats
    size_t inter_off = 128 + (size_t)2 * NEXP * T_TOKENS * 4;       // 16B aligned
    bf16*  inter   = (bf16*)(ws + inter_off);                       // <= 9216*4096 bf16 (~72MB)

    (void)hipMemsetAsync(counts, 0, 64, stream);
    (void)hipMemsetAsync(out, 0, (size_t)T_TOKENS * H_DIM * sizeof(float), stream);

    router_kernel<<<T_TOKENS / 8, 256, 0, stream>>>(x, router_w, counts, tok_idx, tok_w);
    scan_kernel<<<1, 32, 0, stream>>>(counts, row_off);

    dim3 gA(NEXP * MAX_MT, I_DIM / NTILE);   // 256 x 32
    expert_in_kernel<<<gA, 256, 0, stream>>>(x, w_in, b_in, counts, row_off, tok_idx, inter);

    dim3 gB(NEXP * MAX_MT, H_DIM / NTILE);   // 256 x 8
    expert_out_kernel<<<gB, 256, 0, stream>>>(inter, w_out, b_out, counts, row_off,
                                              tok_idx, tok_w, out);
}
